// MutiHeadAttention_89335319757600
// MI455X (gfx1250) — matrix operand order repaired and checked by reading, not compiled, not run on hardware
//
#include <hip/hip_runtime.h>

typedef __attribute__((ext_vector_type(16))) __bf16 v16bf;
typedef __attribute__((ext_vector_type(8)))  __bf16 v8bf;
typedef __attribute__((ext_vector_type(2)))  __bf16 v2bf;
typedef __attribute__((ext_vector_type(8)))  float  v8f;

#define B_  4
#define S_  2048
#define D_  1024
#define H_  16
#define HD_ 64

union BF16x16 { v16bf v; v8bf h[2]; };

__device__ __forceinline__ v16bf join8(v8bf lo, v8bf hi) {
  BF16x16 u; u.h[0] = lo; u.h[1] = hi; return u.v;
}

__device__ __forceinline__ v16bf loadAB(const __bf16* p) {  // two b128: p+0..7 and p+16..23
  return join8(*(const v8bf*)p, *(const v8bf*)(p + 16));
}
__device__ __forceinline__ v16bf loadAfrag(const __bf16* row, int koff) {
  // A-frag: lanes 0-15 hold K=koff+0..7 and +16..23; lanes 16-31 hold +8..15 and +24..31
  return join8(*(const v8bf*)(row + koff), *(const v8bf*)(row + koff + 16));
}

// ---------------- f32 -> bf16 conversion (vectorized, grid-stride) ----------------
__global__ void cvt_f32_bf16_kernel(const float* __restrict__ src,
                                    __bf16* __restrict__ dst, int n2) {
  int i = blockIdx.x * blockDim.x + threadIdx.x;
  int stride = gridDim.x * blockDim.x;
  const float2* s2 = (const float2*)src;
  v2bf* d2 = (v2bf*)dst;
  for (; i < n2; i += stride) {
    float2 f = s2[i];
    v2bf o; o[0] = (__bf16)f.x; o[1] = (__bf16)f.y;
    d2[i] = o;
  }
}

// ---------------- fused QKV projection: q/k/v = x @ W^T + b ----------------
// One wave computes a 32x64 output tile (2 M-subtiles x 4 N-subtiles), software
// pipelined (double-buffered fragments). Q gets 0.125*log2(e) folded in.
// V is stored transposed [B,H,hd,S] so the attention PV-WMMA loads contiguously.
__global__ void qkv_proj_kernel(const __bf16* __restrict__ xb,
                                const __bf16* __restrict__ wqb,
                                const __bf16* __restrict__ wkb,
                                const __bf16* __restrict__ wvb,
                                const float* __restrict__ bq,
                                const float* __restrict__ bk,
                                const float* __restrict__ bv,
                                __bf16* __restrict__ Qb,
                                __bf16* __restrict__ Kb,
                                __bf16* __restrict__ Vtb) {
  const int lane = threadIdx.x & 31;
  const int w    = (blockIdx.x * blockDim.x + threadIdx.x) >> 5;
  const int nt   = w & 15;          // 16 col tiles of 64
  const int mt   = (w >> 4) & 255;  // 256 row tiles of 32
  const int qkv  = w >> 12;         // 0=Q 1=K 2=V
  const int half = lane >> 4;
  const int l16  = lane & 15;

  const __bf16* wmat = (qkv == 0) ? wqb : (qkv == 1) ? wkb : wvb;
  const float*  bias = (qkv == 0) ? bq  : (qkv == 1) ? bk  : bv;

  const __bf16* arow0 = xb + (size_t)(mt * 32 + l16) * D_;
  const __bf16* arow1 = arow0 + (size_t)16 * D_;
  const __bf16* brow[4];
#pragma unroll
  for (int j = 0; j < 4; ++j)
    brow[j] = wmat + (size_t)(nt * 64 + j * 16 + l16) * D_;

  v8f acc[2][4] = {};

  // prologue: fragments for kk = 0
  const int aoff = half * 8;    // + kk
  const int boff = half * 8;    // + kk
  v16bf ac[2], bc[4];
  ac[0] = loadAfrag(arow0, aoff);
  ac[1] = loadAfrag(arow1, aoff);
#pragma unroll
  for (int j = 0; j < 4; ++j) bc[j] = loadAB(brow[j] + boff);

  for (int kk = 0; kk < D_; kk += 32) {
    const int nk = (kk + 32 < D_) ? kk + 32 : 0;  // last step reloads k=0 (discarded)
    if (kk + 128 < D_) __builtin_prefetch(arow0 + kk + 128, 0, 1);
    // issue next-step loads before consuming current fragments
    v16bf an[2], bn[4];
    an[0] = loadAfrag(arow0, nk + aoff);
    an[1] = loadAfrag(arow1, nk + aoff);
#pragma unroll
    for (int j = 0; j < 4; ++j) bn[j] = loadAB(brow[j] + nk + boff);
#pragma unroll
    for (int j = 0; j < 4; ++j) {
      acc[0][j] = __builtin_amdgcn_wmma_f32_16x16x32_bf16(false, ac[0], false, bc[j],
                                                          (short)0, acc[0][j], false, false);
      acc[1][j] = __builtin_amdgcn_wmma_f32_16x16x32_bf16(false, ac[1], false, bc[j],
                                                          (short)0, acc[1][j], false, false);
    }
    ac[0] = an[0]; ac[1] = an[1];
#pragma unroll
    for (int j = 0; j < 4; ++j) bc[j] = bn[j];
  }

  const float qscale = 0.125f * 1.4426950408889634f;  // 1/sqrt(hd) * log2(e)
#pragma unroll
  for (int j = 0; j < 4; ++j) {
    const int ncol = nt * 64 + j * 16 + l16;
    const float bb = bias[ncol];
#pragma unroll
    for (int mi = 0; mi < 2; ++mi) {
#pragma unroll
      for (int r = 0; r < 8; ++r) {
        const int gm = mt * 32 + mi * 16 + r + half * 8;  // D-layout: lanes16-31 = M+8
        const float val = acc[mi][j][r] + bb;
        if (qkv == 0) {
          Qb[(size_t)gm * D_ + ncol] = (__bf16)(val * qscale);
        } else if (qkv == 1) {
          Kb[(size_t)gm * D_ + ncol] = (__bf16)val;
        } else {
          const int b  = gm >> 11, s  = gm & 2047;
          const int h  = ncol >> 6, dh = ncol & 63;
          Vtb[((size_t)(b * H_ + h) * HD_ + dh) * S_ + s] = (__bf16)val;
        }
      }
    }
  }
}

// ---------------- flash attention: one wave per (b, h, 16-query tile) ----------------
__global__ void attn_kernel(const __bf16* __restrict__ Qb,
                            const __bf16* __restrict__ Kb,
                            const __bf16* __restrict__ Vtb,
                            __bf16* __restrict__ Ab) {
  __shared__ __align__(16) __bf16 pbuf[8][16][40];  // per-wave 16x32 P staging (padded)
  const int lane = threadIdx.x & 31;
  const int wib  = threadIdx.x >> 5;
  const int w    = (blockIdx.x * blockDim.x + threadIdx.x) >> 5;
  const int qt   = w & 127;
  const int h    = (w >> 7) & 15;
  const int b    = w >> 11;
  const int half = lane >> 4;
  const int l16  = lane & 15;

  // Q tile A-frags (16x64 -> two 16x32 frags), row = lane&15
  const __bf16* qrow = Qb + ((size_t)b * S_ + qt * 16 + l16) * D_ + h * HD_;
  const v16bf a0 = loadAfrag(qrow, half * 8);
  const v16bf a1 = loadAfrag(qrow + 32, half * 8);

  const __bf16* kbase = Kb + (size_t)b * S_ * D_ + h * HD_;
  const __bf16* vbase = Vtb + (size_t)(b * H_ + h) * HD_ * S_;

  v8f o[4] = {};
  float m[8], l[8];
#pragma unroll
  for (int r = 0; r < 8; ++r) { m[r] = -1e30f; l[r] = 0.0f; }

  for (int kk = 0; kk < S_; kk += 32) {
    // ---- issue ALL global loads for this chunk up front (K frags + V frags) ----
    const __bf16* k0 = kbase + (size_t)(kk + l16) * D_ + half * 8;
    const __bf16* k1 = kbase + (size_t)(kk + 16 + l16) * D_ + half * 8;
    v16bf bk00 = loadAB(k0);
    v16bf bk01 = loadAB(k0 + 32);
    v16bf bk10 = loadAB(k1);
    v16bf bk11 = loadAB(k1 + 32);
    v16bf bv[4];
#pragma unroll
    for (int dt = 0; dt < 4; ++dt)  // V latency hides behind score WMMAs + softmax
      bv[dt] = loadAB(vbase + (size_t)(dt * 16 + l16) * S_ + kk + half * 8);

    // ---- scores: S = Q K^T (two 16x16 tiles, K-dim = hd=64 in 2 steps) ----
    v8f z = {};
    v8f s0 = __builtin_amdgcn_wmma_f32_16x16x32_bf16(false, a0, false, bk00, (short)0, z,  false, false);
    s0     = __builtin_amdgcn_wmma_f32_16x16x32_bf16(false, a1, false, bk01, (short)0, s0, false, false);
    v8f s1 = __builtin_amdgcn_wmma_f32_16x16x32_bf16(false, a0, false, bk10, (short)0, z,  false, false);
    s1     = __builtin_amdgcn_wmma_f32_16x16x32_bf16(false, a1, false, bk11, (short)0, s1, false, false);

    // ---- online softmax (scores pre-scaled by 1/sqrt(hd)*log2e, use exp2) ----
    float p0[8], p1[8];
#pragma unroll
    for (int r = 0; r < 8; ++r) {
      float v = fmaxf(s0[r], s1[r]);
      v = fmaxf(v, __shfl_xor(v, 1, 32));
      v = fmaxf(v, __shfl_xor(v, 2, 32));
      v = fmaxf(v, __shfl_xor(v, 4, 32));
      v = fmaxf(v, __shfl_xor(v, 8, 32));
      const float mnew  = fmaxf(m[r], v);
      const float alpha = exp2f(m[r] - mnew);
      p0[r] = exp2f(s0[r] - mnew);
      p1[r] = exp2f(s1[r] - mnew);
      float ps = p0[r] + p1[r];
      ps += __shfl_xor(ps, 1, 32);
      ps += __shfl_xor(ps, 2, 32);
      ps += __shfl_xor(ps, 4, 32);
      ps += __shfl_xor(ps, 8, 32);
      l[r] = l[r] * alpha + ps;
      m[r] = mnew;
      o[0][r] *= alpha; o[1][r] *= alpha; o[2][r] *= alpha; o[3][r] *= alpha;
    }

    // ---- transpose P from D-layout to A-layout through per-wave LDS slice ----
    // LDS ops from one wave are processed in order by HW (DScnt in-order), and
    // pbuf[wib] is private to this wave, so only compiler-level ordering is
    // needed -> wave_barrier instead of full workgroup s_barrier.
    __builtin_amdgcn_wave_barrier();   // WAR: prev iteration's reads before these writes
#pragma unroll
    for (int r = 0; r < 8; ++r) {
      const int prow = r + half * 8;
      pbuf[wib][prow][l16]      = (__bf16)p0[r];
      pbuf[wib][prow][16 + l16] = (__bf16)p1[r];
    }
    __builtin_amdgcn_wave_barrier();   // RAW: writes before cross-lane reads
    const __bf16* pr = &pbuf[wib][l16][0];
    const v16bf pa = join8(*(const v8bf*)(pr + half * 8),
                           *(const v8bf*)(pr + half * 8 + 16));

    // ---- O += P V (V^T stored so per-lane loads are contiguous) ----
#pragma unroll
    for (int dt = 0; dt < 4; ++dt) {
      o[dt] = __builtin_amdgcn_wmma_f32_16x16x32_bf16(false, pa, false, bv[dt],
                                                      (short)0, o[dt], false, false);
    }
  }

  // ---- normalize and store concat-head layout [B*S, D] ----
#pragma unroll
  for (int r = 0; r < 8; ++r) {
    const float inv = 1.0f / l[r];
    const int tok = b * S_ + qt * 16 + r + half * 8;
#pragma unroll
    for (int dt = 0; dt < 4; ++dt) {
      const int col = h * HD_ + dt * 16 + l16;
      Ab[(size_t)tok * D_ + col] = (__bf16)(o[dt][r] * inv);
    }
  }
}

// ---------------- output projection: out = attn @ Wo^T + bo (f32 out) ----------------
// Same 32x64 pipelined tile as the QKV GEMM.
__global__ void out_proj_kernel(const __bf16* __restrict__ Ab,
                                const __bf16* __restrict__ wob,
                                const float* __restrict__ bo,
                                float* __restrict__ out) {
  const int lane = threadIdx.x & 31;
  const int w    = (blockIdx.x * blockDim.x + threadIdx.x) >> 5;
  const int nt   = w & 15;
  const int mt   = w >> 4;  // 0..255, tiles of 32 rows
  const int half = lane >> 4;
  const int l16  = lane & 15;

  const __bf16* arow0 = Ab + (size_t)(mt * 32 + l16) * D_;
  const __bf16* arow1 = arow0 + (size_t)16 * D_;
  const __bf16* brow[4];
#pragma unroll
  for (int j = 0; j < 4; ++j)
    brow[j] = wob + (size_t)(nt * 64 + j * 16 + l16) * D_;

  v8f acc[2][4] = {};
  const int aoff = half * 8;
  const int boff = half * 8;
  v16bf ac[2], bc[4];
  ac[0] = loadAfrag(arow0, aoff);
  ac[1] = loadAfrag(arow1, aoff);
#pragma unroll
  for (int j = 0; j < 4; ++j) bc[j] = loadAB(brow[j] + boff);

  for (int kk = 0; kk < D_; kk += 32) {
    const int nk = (kk + 32 < D_) ? kk + 32 : 0;
    if (kk + 128 < D_) __builtin_prefetch(arow0 + kk + 128, 0, 1);
    v16bf an[2], bn[4];
    an[0] = loadAfrag(arow0, nk + aoff);
    an[1] = loadAfrag(arow1, nk + aoff);
#pragma unroll
    for (int j = 0; j < 4; ++j) bn[j] = loadAB(brow[j] + nk + boff);
#pragma unroll
    for (int j = 0; j < 4; ++j) {
      acc[0][j] = __builtin_amdgcn_wmma_f32_16x16x32_bf16(false, ac[0], false, bc[j],
                                                          (short)0, acc[0][j], false, false);
      acc[1][j] = __builtin_amdgcn_wmma_f32_16x16x32_bf16(false, ac[1], false, bc[j],
                                                          (short)0, acc[1][j], false, false);
    }
    ac[0] = an[0]; ac[1] = an[1];
#pragma unroll
    for (int j = 0; j < 4; ++j) bc[j] = bn[j];
  }

#pragma unroll
  for (int j = 0; j < 4; ++j) {
    const int ncol = nt * 64 + j * 16 + l16;
    const float bb = bo[ncol];
#pragma unroll
    for (int mi = 0; mi < 2; ++mi) {
#pragma unroll
      for (int r = 0; r < 8; ++r) {
        const int gm = mt * 32 + mi * 16 + r + half * 8;
        out[(size_t)gm * D_ + ncol] = acc[mi][j][r] + bb;
      }
    }
  }
}

extern "C" void kernel_launch(void* const* d_in, const int* in_sizes, int n_in,
                              void* d_out, int out_size, void* d_ws, size_t ws_size,
                              hipStream_t stream) {
  const float* x  = (const float*)d_in[0];
  const float* wq = (const float*)d_in[1];
  const float* bq = (const float*)d_in[2];
  const float* wk = (const float*)d_in[3];
  const float* bk = (const float*)d_in[4];
  const float* wv = (const float*)d_in[5];
  const float* bv = (const float*)d_in[6];
  const float* wo = (const float*)d_in[7];
  const float* bo = (const float*)d_in[8];
  float* out = (float*)d_out;

  const size_t NX = (size_t)B_ * S_ * D_;   // 8388608
  const size_t NW = (size_t)D_ * D_;        // 1048576

  __bf16* xb  = (__bf16*)d_ws;
  __bf16* wqb = xb  + NX;
  __bf16* wkb = wqb + NW;
  __bf16* wvb = wkb + NW;
  __bf16* wob = wvb + NW;
  __bf16* Qb  = wob + NW;
  __bf16* Kb  = Qb  + NX;
  __bf16* Vtb = Kb  + NX;   // [B,H,hd,S]
  __bf16* Ab  = Vtb + NX;   // concat-head attention output

  cvt_f32_bf16_kernel<<<2048, 256, 0, stream>>>(x,  xb,  (int)(NX / 2));
  cvt_f32_bf16_kernel<<<512,  256, 0, stream>>>(wq, wqb, (int)(NW / 2));
  cvt_f32_bf16_kernel<<<512,  256, 0, stream>>>(wk, wkb, (int)(NW / 2));
  cvt_f32_bf16_kernel<<<512,  256, 0, stream>>>(wv, wvb, (int)(NW / 2));
  cvt_f32_bf16_kernel<<<512,  256, 0, stream>>>(wo, wob, (int)(NW / 2));

  // 3 matrices * 256 M-tiles * 16 N-tiles = 12288 waves -> 1536 blocks of 256
  qkv_proj_kernel<<<1536, 256, 0, stream>>>(xb, wqb, wkb, wvb, bq, bk, bv, Qb, Kb, Vtb);
  // 4*16*128 = 8192 waves -> 1024 blocks of 256
  attn_kernel<<<1024, 256, 0, stream>>>(Qb, Kb, Vtb, Ab);
  // 256*16 = 4096 waves -> 512 blocks of 256
  out_proj_kernel<<<512, 256, 0, stream>>>(Ab, wob, bo, out);
}
